// EncoderLayer_1580547975430
// MI455X (gfx1250) — compile-verified, matrix operand order checked by reading, not run on hardware
//
#include <hip/hip_runtime.h>
#include <hip/hip_bf16.h>

// ---------------------------------------------------------------------------
// CDNA5 (gfx1250) transformer encoder layer, all GEMMs on
// v_wmma_f32_16x16x32_bf16 (bf16 in, f32 accum).
//
//  * reference's reshape(B*H, L, DPH) is a raw view -> q/k/v flat buffers are
//    64 contiguous [1024 x 64] row-major matrices; no permutes anywhere.
//  * ALL GEMM B-operands are B^T row-major [N,K] bf16 (weights transposed
//    during f32->bf16 conversion; V via a tiny batched bf16 transpose), so
//    every fragment load is K-contiguous dwords (merged into b128 clauses).
//  * Tile BM=128, BN=128, BK=64; 512 threads = 16 waves, 32x32 wave tile:
//    keeps per-wave VGPRs ~80 (no scratch spills) while amortizing the A-tile
//    stage over 128 columns.
//  * fp32 attention scores go straight into d_out, softmax in place; the
//    context GEMM converts them to bf16 while staging to LDS.
//  * Prefetches are unconditional: speculative prefetch drops invalid addrs.
// ---------------------------------------------------------------------------

typedef __attribute__((ext_vector_type(16))) __bf16 v16bf;
typedef __attribute__((ext_vector_type(8)))  float  v8f;

static __device__ __forceinline__ unsigned short f2bf(float f) {
  // round-to-nearest-even fp32 -> bf16
  unsigned int x = __float_as_uint(f);
  unsigned int r = x + 0x7fffu + ((x >> 16) & 1u);
  return (unsigned short)(r >> 16);
}

union FragB {
  v16bf v;
  unsigned int   u32[8];
  unsigned short u16[16];
};

// ---------------------------------------------------------------------------
// WMMA GEMM:  C = alpha * (A @ Bt^T) + bias, optional ReLU.
//   A : row-major [M,K]; bf16 (A_F32=0) or fp32 converted while staging (=1)
//   Bt: row-major [N,K] bf16 -> K-contiguous dword fragment loads
//   Tile: BM=128, BN=128, BK=64.  512 threads = 16 waves (4x4), 32x32/wave.
//   Batched via blockIdx.z with element strides sA/sB/sC.
// ---------------------------------------------------------------------------
template<int A_F32, int RELU>
__global__ __launch_bounds__(512)
void gemm_wmma(const void* __restrict__ Ap, const unsigned short* __restrict__ Btp,
               const float* __restrict__ bias, float* __restrict__ Cf,
               unsigned short* __restrict__ Cb, int M, int N, int K, float alpha,
               long long sA, long long sB, long long sC)
{
  __shared__ unsigned short Asm[128 * 72];   // BK=64 padded to 72 (144B rows)

  const int tid  = threadIdx.x;
  const int lane = tid & 31;
  const int wid  = tid >> 5;
  const int wm   = wid >> 2;                 // 0..3 -> row offset wm*32
  const int wn   = wid & 3;                  // 0..3 -> col offset wn*32
  const int m0   = blockIdx.y * 128;
  const int n0   = blockIdx.x * 128;
  const long long bz = blockIdx.z;

  const float*          Af = (const float*)Ap          + (A_F32 ? bz * sA : 0);
  const unsigned short* Ab = (const unsigned short*)Ap + (A_F32 ? 0 : bz * sA);
  const unsigned short* Bt = Btp + bz * sB;

  v8f acc[2][2];
  for (int i = 0; i < 2; ++i)
    for (int j = 0; j < 2; ++j)
      for (int r = 0; r < 8; ++r) acc[i][j][r] = 0.0f;

  const int g    = lane >> 4;   // K-half select (lanes 16-31 carry K+8 / K+24)
  const int ln16 = lane & 15;

  for (int k0 = 0; k0 < K; k0 += 64) {
    __syncthreads();
    if (A_F32) {
      // stage 128x64 fp32 -> bf16: 4 float4 loads buffered, then LDS stores
      float4 tmp[4];
      for (int i = 0; i < 4; ++i) {
        int lin = tid + i * 512;                     // 2048 float4 chunks
        int row = lin >> 4, c4 = lin & 15;
        const float* gp = Af + (long long)(m0 + row) * K + k0 + c4 * 4;
        tmp[i] = *(const float4*)gp;
        __builtin_prefetch(gp + 64, 0, 3);           // speculative, OOB dropped
      }
      for (int i = 0; i < 4; ++i) {
        int lin = tid + i * 512;
        int row = lin >> 4, c4 = lin & 15;
        uint2 pk;
        pk.x = (unsigned int)f2bf(tmp[i].x) | ((unsigned int)f2bf(tmp[i].y) << 16);
        pk.y = (unsigned int)f2bf(tmp[i].z) | ((unsigned int)f2bf(tmp[i].w) << 16);
        *(uint2*)(&Asm[row * 72 + c4 * 4]) = pk;
      }
    } else {
      // stage 128x64 bf16: 2 uint4 (8 bf16) loads buffered, then LDS stores
      uint4 tmp[2];
      for (int i = 0; i < 2; ++i) {
        int lin = tid + i * 512;                     // 1024 uint4 chunks
        int row = lin >> 3, c8 = lin & 7;
        const unsigned short* gp = Ab + (long long)(m0 + row) * K + k0 + c8 * 8;
        tmp[i] = *(const uint4*)gp;
        __builtin_prefetch(gp + 64, 0, 3);           // speculative, OOB dropped
      }
      for (int i = 0; i < 2; ++i) {
        int lin = tid + i * 512;
        int row = lin >> 3, c8 = lin & 7;
        *(uint4*)(&Asm[row * 72 + c8 * 8]) = tmp[i];
      }
    }
    __syncthreads();

#pragma unroll
    for (int ks = 0; ks < 2; ++ks) {
      FragB a[2], b[2];
      // A fragments from LDS (16-bit A 16x32 layout: lane=M, pair K-map)
      for (int t = 0; t < 2; ++t) {
        int m = wm * 32 + t * 16 + ln16;
#pragma unroll
        for (int v = 0; v < 8; ++v) {
          int kk = (v < 4) ? (2 * v + 8 * g) : (2 * v + 8 + 8 * g);
          a[t].u32[v] = *(const unsigned int*)(&Asm[m * 72 + ks * 32 + kk]);
        }
      }
      // B fragments: K-contiguous dwords from B^T rows (L2-resident)
      for (int t = 0; t < 2; ++t) {
        int n = n0 + wn * 32 + t * 16 + ln16;
        if (n >= N) n = N - 1;                       // safe for N=64 tails
        const unsigned short* bp = Bt + (long long)n * K + k0 + ks * 32;
#pragma unroll
        for (int v = 0; v < 8; ++v) {
          int kk = (v < 4) ? (2 * v + 8 * g) : (2 * v + 8 + 8 * g);
          b[t].u32[v] = *(const unsigned int*)(bp + kk);
        }
      }
#pragma unroll
      for (int tm = 0; tm < 2; ++tm)
#pragma unroll
        for (int tn = 0; tn < 2; ++tn)
          acc[tm][tn] = __builtin_amdgcn_wmma_f32_16x16x32_bf16(
              false, a[tm].v, false, b[tn].v, (short)0, acc[tm][tn], false, false);
    }
  }

  float*          Cfp = Cf ? Cf + bz * sC : nullptr;
  unsigned short* Cbp = Cb ? Cb + bz * sC : nullptr;
  for (int tm = 0; tm < 2; ++tm)
    for (int tn = 0; tn < 2; ++tn)
      for (int r = 0; r < 8; ++r) {
        int m = m0 + wm * 32 + tm * 16 + r + g * 8;   // C layout: VGPR r, lane half
        int n = n0 + wn * 32 + tn * 16 + ln16;
        if (m < M && n < N) {
          float val = acc[tm][tn][r] * alpha;
          if (bias) val += bias[n];
          if (RELU) val = fmaxf(val, 0.0f);
          long long idx = (long long)m * N + n;
          if (Cfp) Cfp[idx] = val;
          if (Cbp) Cbp[idx] = f2bf(val);
        }
      }
}

// ---------------------------------------------------------------------------
// Row softmax in place, row length 1024, one 256-thread (8-wave) block per row.
// ---------------------------------------------------------------------------
__global__ __launch_bounds__(256)
void softmax_rows(float* __restrict__ p)
{
  __shared__ float red[256];
  const long long base = (long long)blockIdx.x * 1024;
  const int t = threadIdx.x;
  float v[4]; float mx = -3.4e38f;
  for (int i = 0; i < 4; ++i) { v[i] = p[base + t + i * 256]; mx = fmaxf(mx, v[i]); }
  red[t] = mx; __syncthreads();
  for (int s = 128; s > 0; s >>= 1) { if (t < s) red[t] = fmaxf(red[t], red[t + s]); __syncthreads(); }
  mx = red[0]; __syncthreads();
  float sum = 0.0f;
  for (int i = 0; i < 4; ++i) { v[i] = expf(v[i] - mx); sum += v[i]; }
  red[t] = sum; __syncthreads();
  for (int s = 128; s > 0; s >>= 1) { if (t < s) red[t] += red[t + s]; __syncthreads(); }
  const float inv = 1.0f / red[0];
  for (int i = 0; i < 4; ++i) p[base + t + i * 256] = v[i] * inv;
}

// ---------------------------------------------------------------------------
// y = LayerNorm(res + x) * g + b, row length 1024; fp32 out + optional bf16 out
// ---------------------------------------------------------------------------
__global__ __launch_bounds__(256)
void add_layernorm(const float* __restrict__ res, const float* __restrict__ x,
                   const float* __restrict__ gam, const float* __restrict__ bet,
                   float* __restrict__ yf, unsigned short* __restrict__ ybf)
{
  __shared__ float red[256];
  const long long base = (long long)blockIdx.x * 1024;
  const int t = threadIdx.x;
  float v[4]; float s = 0.0f;
  for (int i = 0; i < 4; ++i) { int c = t + i * 256; v[i] = res[base + c] + x[base + c]; s += v[i]; }
  red[t] = s; __syncthreads();
  for (int q = 128; q > 0; q >>= 1) { if (t < q) red[t] += red[t + q]; __syncthreads(); }
  const float mu = red[0] * (1.0f / 1024.0f); __syncthreads();
  float s2 = 0.0f;
  for (int i = 0; i < 4; ++i) { float d = v[i] - mu; s2 += d * d; }
  red[t] = s2; __syncthreads();
  for (int q = 128; q > 0; q >>= 1) { if (t < q) red[t] += red[t + q]; __syncthreads(); }
  const float r = rsqrtf(red[0] * (1.0f / 1024.0f) + 1e-5f);
  for (int i = 0; i < 4; ++i) {
    int c = t + i * 256;
    float y = (v[i] - mu) * r * gam[c] + bet[c];
    yf[base + c] = y;
    if (ybf) ybf[base + c] = f2bf(y);
  }
}

// fp32 -> bf16, flat
__global__ __launch_bounds__(256)
void cvt_f32_bf16(const float* __restrict__ x, unsigned short* __restrict__ y, long long n)
{
  long long i = (long long)blockIdx.x * blockDim.x + threadIdx.x;
  const long long stride = (long long)gridDim.x * blockDim.x;
  for (; i < n; i += stride) y[i] = f2bf(x[i]);
}

// fp32 [R,C] -> bf16 transposed [C,R], LDS-tiled 32x32, blockDim (32,8)
__global__ __launch_bounds__(256)
void cvt_tr_bf16(const float* __restrict__ x, unsigned short* __restrict__ y, int R, int C)
{
  __shared__ float t[32][33];
  const int bx = blockIdx.x * 32, by = blockIdx.y * 32;
  const int tx = threadIdx.x, ty = threadIdx.y;
  for (int i = 0; i < 32; i += 8)
    t[ty + i][tx] = x[(long long)(by + ty + i) * C + bx + tx];
  __syncthreads();
  for (int i = 0; i < 32; i += 8)
    y[(long long)(bx + ty + i) * R + by + tx] = f2bf(t[tx][ty + i]);
}

// bf16 [R,C] -> bf16 transposed [C,R], batched (blockIdx.z, stride sz elems)
__global__ __launch_bounds__(256)
void tr_bf16(const unsigned short* __restrict__ x, unsigned short* __restrict__ y,
             int R, int C, long long sz)
{
  __shared__ unsigned short t[32][33];
  const long long base = (long long)blockIdx.z * sz;
  const int bx = blockIdx.x * 32, by = blockIdx.y * 32;
  const int tx = threadIdx.x, ty = threadIdx.y;
  for (int i = 0; i < 32; i += 8)
    t[ty + i][tx] = x[base + (long long)(by + ty + i) * C + bx + tx];
  __syncthreads();
  for (int i = 0; i < 32; i += 8)
    y[base + (long long)(bx + ty + i) * R + by + tx] = t[tx][ty + i];
}

// ---------------------------------------------------------------------------
extern "C" void kernel_launch(void* const* d_in, const int* in_sizes, int n_in,
                              void* d_out, int out_size, void* d_ws, size_t ws_size,
                              hipStream_t stream)
{
  (void)in_sizes; (void)n_in; (void)out_size; (void)ws_size;
  constexpr int B = 4, L = 1024, D = 1024, H = 16, FFN = 4096;
  constexpr int M   = B * L;     // 4096 activation rows
  constexpr int BH  = B * H;     // 64 raw-view attention chunks
  constexpr int DPH = D / H;     // 64

  const float* inp = (const float*)d_in[0];
  const float* Wq  = (const float*)d_in[1];  const float* bq  = (const float*)d_in[2];
  const float* Wk  = (const float*)d_in[3];  const float* bk  = (const float*)d_in[4];
  const float* Wv  = (const float*)d_in[5];  const float* bv  = (const float*)d_in[6];
  const float* Wo  = (const float*)d_in[7];  const float* bo  = (const float*)d_in[8];
  const float* g1  = (const float*)d_in[9];  const float* be1 = (const float*)d_in[10];
  const float* W1  = (const float*)d_in[11]; const float* b1  = (const float*)d_in[12];
  const float* W2  = (const float*)d_in[13]; const float* b2  = (const float*)d_in[14];
  const float* g2  = (const float*)d_in[15]; const float* be2 = (const float*)d_in[16];

  float* outf = (float*)d_out;                       // [M, D]
  float* attn = outf + (long long)M * D;             // [BH, L, L], softmax'd in place

  char* ws = (char*)d_ws;
  size_t off = 0;
  auto alloc = [&](size_t bytes) -> char* {
    char* p = ws + off;
    off += (bytes + 255) & ~(size_t)255;
    return p;
  };
  unsigned short* Xb   = (unsigned short*)alloc((size_t)M * D * 2);
  unsigned short* Wqt  = (unsigned short*)alloc((size_t)D * D * 2);     // [N,K]
  unsigned short* Wkt  = (unsigned short*)alloc((size_t)D * D * 2);
  unsigned short* Wvt  = (unsigned short*)alloc((size_t)D * D * 2);
  unsigned short* Wot  = (unsigned short*)alloc((size_t)D * D * 2);
  unsigned short* W1t  = (unsigned short*)alloc((size_t)D * FFN * 2);   // [4096,1024]
  unsigned short* W2t  = (unsigned short*)alloc((size_t)FFN * D * 2);   // [1024,4096]
  unsigned short* Qb   = (unsigned short*)alloc((size_t)M * D * 2);
  unsigned short* Kb   = (unsigned short*)alloc((size_t)M * D * 2);
  unsigned short* Vb   = (unsigned short*)alloc((size_t)M * D * 2);
  unsigned short* Vtb  = (unsigned short*)alloc((size_t)M * D * 2);     // per-bh [64,1024]
  unsigned short* Cxb  = (unsigned short*)alloc((size_t)M * D * 2);
  unsigned short* LN1b = (unsigned short*)alloc((size_t)M * D * 2);
  unsigned short* Hb   = (unsigned short*)alloc((size_t)M * FFN * 2);
  float*          T0   = (float*)alloc((size_t)M * D * 4);   // reused for FFN2 out
  float*          LN1f = (float*)alloc((size_t)M * D * 4);

  const dim3 blk(256);
  const dim3 gblk(512);
  const dim3 tblk(32, 8);

  // X: flat f32->bf16; weights: convert + transpose to [N,K]
  cvt_f32_bf16<<<2048, blk, 0, stream>>>(inp, Xb, (long long)M * D);
  cvt_tr_bf16<<<dim3(32, 32), tblk, 0, stream>>>(Wq, Wqt, D, D);
  cvt_tr_bf16<<<dim3(32, 32), tblk, 0, stream>>>(Wk, Wkt, D, D);
  cvt_tr_bf16<<<dim3(32, 32), tblk, 0, stream>>>(Wv, Wvt, D, D);
  cvt_tr_bf16<<<dim3(32, 32), tblk, 0, stream>>>(Wo, Wot, D, D);
  cvt_tr_bf16<<<dim3(FFN / 32, 32), tblk, 0, stream>>>(W1, W1t, D, FFN);
  cvt_tr_bf16<<<dim3(32, FFN / 32), tblk, 0, stream>>>(W2, W2t, FFN, D);

  // QKV projections: [4096,1024] @ W^T[1024,1024] + bias -> bf16
  const dim3 gP(D / 128, M / 128, 1);
  gemm_wmma<0, 0><<<gP, gblk, 0, stream>>>(Xb, Wqt, bq, nullptr, Qb, M, D, D, 1.0f, 0, 0, 0);
  gemm_wmma<0, 0><<<gP, gblk, 0, stream>>>(Xb, Wkt, bk, nullptr, Kb, M, D, D, 1.0f, 0, 0, 0);
  gemm_wmma<0, 0><<<gP, gblk, 0, stream>>>(Xb, Wvt, bv, nullptr, Vb, M, D, D, 1.0f, 0, 0, 0);

  // Scores: 64 batched [1024,64] x [1024,64]^T (K-matrix IS B^T row-major),
  // alpha = DPH^-0.5, fp32 straight into the d_out attention region.
  const dim3 gS(L / 128, L / 128, BH);
  gemm_wmma<0, 0><<<gS, gblk, 0, stream>>>(Qb, Kb, nullptr, attn, nullptr, L, L, DPH, 0.125f,
                                           (long long)L * DPH, (long long)L * DPH,
                                           (long long)L * L);

  softmax_rows<<<BH * L, blk, 0, stream>>>(attn);

  // V^T per chunk: [1024,64] -> [64,1024] so context uses dword B loads too
  tr_bf16<<<dim3(2, 32, BH), tblk, 0, stream>>>(Vb, Vtb, L, DPH, (long long)L * DPH);

  // Context: 64 batched [1024,1024](fp32 probs, bf16-converted while staging)
  // @ Vt^T -> bf16 context (flat buffer is already [M, D] row-major)
  const dim3 gC(1, L / 128, BH);
  gemm_wmma<1, 0><<<gC, gblk, 0, stream>>>(attn, Vtb, nullptr, nullptr, Cxb, L, DPH, L, 1.0f,
                                           (long long)L * L, (long long)L * DPH,
                                           (long long)L * DPH);

  // Output projection
  gemm_wmma<0, 0><<<gP, gblk, 0, stream>>>(Cxb, Wot, bo, T0, nullptr, M, D, D, 1.0f, 0, 0, 0);

  // ln1 = LN(inputs + attn_out): fp32 for residual/LN2, bf16 for FFN GEMM
  add_layernorm<<<M, blk, 0, stream>>>(inp, T0, g1, be1, LN1f, LN1b);

  // FFN: relu(ln1 @ W1 + b1) @ W2 + b2
  const dim3 gF1(FFN / 128, M / 128, 1);
  gemm_wmma<0, 1><<<gF1, gblk, 0, stream>>>(LN1b, W1t, b1, nullptr, Hb, M, FFN, D, 1.0f, 0, 0, 0);
  const dim3 gF2(D / 128, M / 128, 1);
  gemm_wmma<0, 0><<<gF2, gblk, 0, stream>>>(Hb, W2t, b2, T0, nullptr, M, D, FFN, 1.0f, 0, 0, 0);

  // final: LN(ln1 + ffn) -> d_out
  add_layernorm<<<M, blk, 0, stream>>>(LN1f, T0, g2, be2, outf, nullptr);
}
